// TransformerBlock_64355789963340
// MI455X (gfx1250) — matrix operand order repaired and checked by reading, not compiled, not run on hardware
//
#include <hip/hip_runtime.h>
#include <hip/hip_bf16.h>
#include <math.h>

// ---------------------------------------------------------------------------
// Types / helpers
// ---------------------------------------------------------------------------
typedef __bf16 bf16x16 __attribute__((ext_vector_type(16)));
typedef float  f32x8   __attribute__((ext_vector_type(8)));

union Frag16 { bf16x16 v; unsigned int u[8]; };

__device__ __forceinline__ unsigned short f2bf(float f) {
  unsigned int u = __float_as_uint(f);
  u += 0x7FFFu + ((u >> 16) & 1u);          // round-to-nearest-even
  return (unsigned short)(u >> 16);
}

__device__ __forceinline__ f32x8 wmma_bf16(const Frag16& a, const Frag16& b, f32x8 c) {
  return __builtin_amdgcn_wmma_f32_16x16x32_bf16(
      /*neg_a=*/false, a.v, /*neg_b=*/false, b.v,
      /*c_mod=*/(short)0, c, /*reuse_a=*/false, /*reuse_b=*/false);
}

// Async global -> LDS copy of 16 bytes (GLOBAL_LOAD_ASYNC_TO_LDS_B128,
// tracked by ASYNCcnt). Bypasses VGPRs entirely.
__device__ __forceinline__ void async_copy_b128(unsigned lds_off, const void* gaddr) {
  asm volatile("global_load_async_to_lds_b128 %0, %1, off"
               :: "v"(lds_off), "v"(gaddr)
               : "memory");
}
__device__ __forceinline__ void wait_asynccnt0() {
  asm volatile("s_wait_asynccnt 0" ::: "memory");
}

// ---------------------------------------------------------------------------
// Weight prep: fp32 [K][N] -> bf16 transposed [N][K] (LDS-tiled, coalesced)
// ---------------------------------------------------------------------------
__global__ __launch_bounds__(256) void transpose_f32_to_bf16(
    const float* __restrict__ in, unsigned short* __restrict__ out,
    int K, int N) {
  __shared__ unsigned short tile[32][33];
  int n0 = blockIdx.x * 32, k0 = blockIdx.y * 32;
  int tx = threadIdx.x, ty = threadIdx.y;  // 32 x 8
#pragma unroll
  for (int i = 0; i < 4; i++)
    tile[ty + i * 8][tx] = f2bf(in[(size_t)(k0 + ty + i * 8) * N + n0 + tx]);
  __syncthreads();
#pragma unroll
  for (int i = 0; i < 4; i++)
    out[(size_t)(n0 + ty + i * 8) * K + k0 + tx] = tile[tx][ty + i * 8];
}

// ---------------------------------------------------------------------------
// LayerNorm over D=768, emits bf16
// ---------------------------------------------------------------------------
#define DM 768
__global__ __launch_bounds__(256) void ln_bf16_kernel(
    const float* __restrict__ x, const float* __restrict__ g,
    const float* __restrict__ b, unsigned short* __restrict__ out) {
  int row = blockIdx.x;
  int tid = threadIdx.x;
  const float* xr = x + (size_t)row * DM;
  float v0 = xr[tid], v1 = xr[tid + 256], v2 = xr[tid + 512];
  __shared__ float red[256];
  red[tid] = v0 + v1 + v2;
  __syncthreads();
  for (int s = 128; s > 0; s >>= 1) {
    if (tid < s) red[tid] += red[tid + s];
    __syncthreads();
  }
  float mu = red[0] * (1.0f / DM);
  __syncthreads();
  float d0 = v0 - mu, d1 = v1 - mu, d2 = v2 - mu;
  red[tid] = d0 * d0 + d1 * d1 + d2 * d2;
  __syncthreads();
  for (int s = 128; s > 0; s >>= 1) {
    if (tid < s) red[tid] += red[tid + s];
    __syncthreads();
  }
  float rstd = rsqrtf(red[0] * (1.0f / DM) + 1e-5f);
  unsigned short* orow = out + (size_t)row * DM;
  orow[tid]       = f2bf(d0 * rstd * g[tid]       + b[tid]);
  orow[tid + 256] = f2bf(d1 * rstd * g[tid + 256] + b[tid + 256]);
  orow[tid + 512] = f2bf(d2 * rstd * g[tid + 512] + b[tid + 512]);
}

// ---------------------------------------------------------------------------
// Tiled WMMA GEMM: C[M,N] = A[M,K](bf16) * W (bf16, pre-transposed [N][K])
//                  (+bias)(+gelu)(+resid)
// Block tile 128x128, K-step 32, 8 waves, each wave a 32x64 tile (2x4 frags).
// Both tiles staged global->LDS with async-to-LDS DMA (no VGPR round trip).
// ---------------------------------------------------------------------------
#define BM 128
#define BN 128
#define BK 32
#define TS_STRIDE 40   // ushorts; 80B row pitch keeps b128 copies aligned

__global__ __launch_bounds__(256) void gemm_bf16_kernel(
    const unsigned short* __restrict__ A,   // [M][K]
    const unsigned short* __restrict__ Bt,  // [N][K]  (W transposed)
    const float* __restrict__ bias, const float* __restrict__ resid,
    float* __restrict__ Cf, unsigned short* __restrict__ Cbf,
    int M, int N, int K, int act) {
  __shared__ unsigned short As[BM * TS_STRIDE];
  __shared__ unsigned short Bs[BN * TS_STRIDE];
  int tid  = threadIdx.x;
  int lane = tid & 31;
  int wave = tid >> 5;
  int n16  = lane & 15, half = lane >> 4;
  int wr = (wave >> 1) * 32, wc = (wave & 1) * 64;
  int bm = blockIdx.y * BM, bn = blockIdx.x * BN;

  f32x8 acc[2][4];
#pragma unroll
  for (int r = 0; r < 2; r++)
#pragma unroll
    for (int c = 0; c < 4; c++)
#pragma unroll
      for (int i = 0; i < 8; i++) acc[r][c][i] = 0.f;

  int r0 = tid >> 2;            // 0..63 : staged row
  int c0 = (tid & 3) * 8;       // 0/8/16/24 : staged k-chunk
  for (int k0 = 0; k0 < K; k0 += BK) {
    // stage A tile [128 x 32] and B tile [128 x 32] via async global->LDS DMA
#pragma unroll
    for (int p = 0; p < 2; p++) {
      int r = p * 64 + r0;
      unsigned lds_a =
          (unsigned)(unsigned long long)&As[r * TS_STRIDE + c0];
      async_copy_b128(lds_a, &A[(size_t)(bm + r) * K + k0 + c0]);
      unsigned lds_b =
          (unsigned)(unsigned long long)&Bs[r * TS_STRIDE + c0];
      async_copy_b128(lds_b, &Bt[(size_t)(bn + r) * K + k0 + c0]);
    }
    // prefetch next K-tile while this one is consumed
    if (k0 + BK < K) {
      __builtin_prefetch(&A[(size_t)(bm + r0) * K + k0 + BK], 0, 3);
      __builtin_prefetch(&Bt[(size_t)(bn + r0) * K + k0 + BK], 0, 3);
    }
    wait_asynccnt0();
    __syncthreads();

    Frag16 af[2], bf[4];
#pragma unroll
    for (int r = 0; r < 2; r++)
#pragma unroll
      for (int j = 0; j < 8; j++) {
        int kk = half * 16 + 2 * j;                      // A-layout (contiguous k)
        af[r].u[j] =
            *(const unsigned int*)&As[(wr + r * 16 + n16) * TS_STRIDE + kk];
      }
#pragma unroll
    for (int c = 0; c < 4; c++)
#pragma unroll
      for (int j = 0; j < 8; j++) {
        int kk = half * 16 + 2 * j;                      // B-layout
        bf[c].u[j] =
            *(const unsigned int*)&Bs[(wc + c * 16 + n16) * TS_STRIDE + kk];
      }
#pragma unroll
    for (int r = 0; r < 2; r++)
#pragma unroll
      for (int c = 0; c < 4; c++)
        acc[r][c] = wmma_bf16(af[r], bf[c], acc[r][c]);
    __syncthreads();
  }

  // epilogue: bias -> gelu(erf) -> residual -> store f32 and/or bf16
#pragma unroll
  for (int r = 0; r < 2; r++)
#pragma unroll
    for (int c = 0; c < 4; c++)
#pragma unroll
      for (int i = 0; i < 8; i++) {
        int row = bm + wr + r * 16 + half * 8 + i;
        int col = bn + wc + c * 16 + n16;
        float v = acc[r][c][i];
        if (bias) v += bias[col];
        if (act == 1) v = 0.5f * v * (1.0f + erff(v * 0.70710678118f));
        if (resid) v += resid[(size_t)row * N + col];
        if (Cf)  Cf[(size_t)row * N + col] = v;
        if (Cbf) Cbf[(size_t)row * N + col] = f2bf(v);
      }
}

// ---------------------------------------------------------------------------
// Flash attention: one wave handles 16 queries of one (b,h).
// S^T = K * Q^T so each lane owns one query's scores; online softmax;
// O = P * V accumulated with WMMA over 32-key chunks.
// ---------------------------------------------------------------------------
#define HEADS 12
#define HD 64
#define SEQ 2048
#define QKV_STRIDE (3 * HEADS * HD)  // 2304

__global__ __launch_bounds__(32) void attn_kernel(
    const unsigned short* __restrict__ qkv, const int* __restrict__ mask,
    unsigned short* __restrict__ out) {
  int blk = blockIdx.x;
  int qt  = blk & 127;                // SEQ/16 = 128
  int h   = (blk >> 7) % HEADS;
  int b   = blk / (128 * HEADS);
  int lane = threadIdx.x;
  int n16 = lane & 15, half = lane >> 4;
  const float scale = 0.125f;         // 1/sqrt(64)

  const unsigned short* qbase = qkv + (size_t)b * SEQ * QKV_STRIDE + h * HD;
  const unsigned short* kbase = qbase + HEADS * HD;
  const unsigned short* vbase = qbase + 2 * HEADS * HD;
  int q0 = qt * 16;

  // Q^T as B-fragments (lane n = query), two 32-dim chunks of hd=64
  Frag16 bq[2];
#pragma unroll
  for (int c = 0; c < 2; c++)
#pragma unroll
    for (int j = 0; j < 8; j++) {
      int kk = c * 32 + half * 16 + 2 * j;
      bq[c].u[j] =
          *(const unsigned int*)&qbase[(size_t)(q0 + n16) * QKV_STRIDE + kk];
    }

  __shared__ float rowmax[16], rowsum[16], alphaS[16];
  __shared__ unsigned short P[16 * 34];
  if (lane < 16) { rowmax[lane] = -1e30f; rowsum[lane] = 0.f; }
  f32x8 o[4];
#pragma unroll
  for (int t = 0; t < 4; t++)
#pragma unroll
    for (int i = 0; i < 8; i++) o[t][i] = 0.f;
  __syncthreads();

  for (int k0 = 0; k0 < SEQ; k0 += 32) {
    // --- scores S^T[32 keys x 16 queries] ---
    f32x8 s[2];
#pragma unroll
    for (int rt = 0; rt < 2; rt++)
#pragma unroll
      for (int i = 0; i < 8; i++) s[rt][i] = 0.f;
#pragma unroll
    for (int rt = 0; rt < 2; rt++)
#pragma unroll
      for (int c = 0; c < 2; c++) {
        Frag16 ak;  // A = K rows (lane m = key)
#pragma unroll
        for (int j = 0; j < 8; j++) {
          int kk = c * 32 + half * 16 + 2 * j;
          ak.u[j] = *(const unsigned int*)
              &kbase[(size_t)(k0 + rt * 16 + n16) * QKV_STRIDE + kk];
        }
        s[rt] = wmma_bf16(ak, bq[c], s[rt]);
      }

    // --- online softmax (per lane: query n16, 16 key-scores) ---
    float sv[16];
    float mx = -1e30f;
#pragma unroll
    for (int rt = 0; rt < 2; rt++)
#pragma unroll
      for (int i = 0; i < 8; i++) {
        int key = k0 + rt * 16 + half * 8 + i;
        float v = s[rt][i] * scale;
        if (mask[b * SEQ + key] == 0) v = -1e30f;
        sv[rt * 8 + i] = v;
        mx = fmaxf(mx, v);
      }
    mx = fmaxf(mx, __shfl_xor(mx, 16, 32));
    if (half == 0) {
      float mo = rowmax[n16];
      float mn = fmaxf(mo, mx);
      rowmax[n16] = mn;
      alphaS[n16] = __expf(mo - mn);
    }
    __syncthreads();
    float mn = rowmax[n16];
    float ts = 0.f;
#pragma unroll
    for (int rt = 0; rt < 2; rt++)
#pragma unroll
      for (int i = 0; i < 8; i++) {
        float p = __expf(sv[rt * 8 + i] - mn);
        ts += p;
        int kin = rt * 16 + half * 8 + i;
        P[n16 * 34 + kin] = f2bf(p);
      }
    ts += __shfl_xor(ts, 16, 32);
    if (half == 0) rowsum[n16] = rowsum[n16] * alphaS[n16] + ts;

    // rescale O accumulators (rows are queries i + 8*half)
    float al[8];
#pragma unroll
    for (int i = 0; i < 8; i++) al[i] = alphaS[half * 8 + i];
#pragma unroll
    for (int t = 0; t < 4; t++)
#pragma unroll
      for (int i = 0; i < 8; i++) o[t][i] *= al[i];
    __syncthreads();

    // --- O += P[16q x 32k] * V[32k x 64d] ---
    Frag16 ap;  // A = P (lane m = query)
#pragma unroll
    for (int j = 0; j < 8; j++) {
      int kk = half * 16 + 2 * j;
      ap.u[j] = *(const unsigned int*)&P[n16 * 34 + kk];
    }
#pragma unroll
    for (int t = 0; t < 4; t++) {
      Frag16 bv;  // B = V chunk (lane n = d within 16-tile)
#pragma unroll
      for (int j = 0; j < 8; j++) {
        int kk = half * 16 + 2 * j;
        unsigned int lo =
            vbase[(size_t)(k0 + kk) * QKV_STRIDE + t * 16 + n16];
        unsigned int hi =
            vbase[(size_t)(k0 + kk + 1) * QKV_STRIDE + t * 16 + n16];
        bv.u[j] = lo | (hi << 16);
      }
      o[t] = wmma_bf16(ap, bv, o[t]);
    }
    __syncthreads();
  }

  float inv[8];
#pragma unroll
  for (int i = 0; i < 8; i++) inv[i] = 1.0f / rowsum[half * 8 + i];
#pragma unroll
  for (int t = 0; t < 4; t++)
#pragma unroll
    for (int i = 0; i < 8; i++) {
      int row = q0 + half * 8 + i;
      int col = h * HD + t * 16 + n16;
      out[((size_t)b * SEQ + row) * (HEADS * HD) + col] = f2bf(o[t][i] * inv[i]);
    }
}

// ---------------------------------------------------------------------------
// Orchestration
// ---------------------------------------------------------------------------
extern "C" void kernel_launch(void* const* d_in, const int* in_sizes, int n_in,
                              void* d_out, int out_size, void* d_ws,
                              size_t ws_size, hipStream_t stream) {
  const float* x     = (const float*)d_in[0];
  const int*   mask  = (const int*)d_in[1];
  const float* ln1_g = (const float*)d_in[2];
  const float* ln1_b = (const float*)d_in[3];
  const float* qkv_w = (const float*)d_in[4];
  const float* qkv_b = (const float*)d_in[5];
  const float* out_w = (const float*)d_in[6];
  const float* out_b = (const float*)d_in[7];
  const float* ln2_g = (const float*)d_in[8];
  const float* ln2_b = (const float*)d_in[9];
  const float* fc1_w = (const float*)d_in[10];
  const float* fc1_b = (const float*)d_in[11];
  const float* fc2_w = (const float*)d_in[12];
  const float* fc2_b = (const float*)d_in[13];
  (void)in_sizes; (void)n_in; (void)out_size; (void)ws_size;

  const int Btok = 2 * 2048;  // 4096 token rows
  const int D = 768, DFF = 3072, NQKV = 2304;

  char* ws = (char*)d_ws;
  size_t off = 0;
  auto take = [&](size_t bytes) -> void* {
    void* p = ws + off;
    off += (bytes + 255) & ~(size_t)255;
    return p;
  };
  unsigned short* qkvw_t = (unsigned short*)take((size_t)D * NQKV * 2);
  unsigned short* outw_t = (unsigned short*)take((size_t)D * D * 2);
  unsigned short* fc1w_t = (unsigned short*)take((size_t)D * DFF * 2);
  unsigned short* fc2w_t = (unsigned short*)take((size_t)DFF * D * 2);
  unsigned short* h1_bf  = (unsigned short*)take((size_t)Btok * D * 2);
  unsigned short* qkv_bf = (unsigned short*)take((size_t)Btok * NQKV * 2);
  unsigned short* attn_bf= (unsigned short*)take((size_t)Btok * D * 2);
  float*          x1     = (float*)take((size_t)Btok * D * 4);
  unsigned short* h2_bf  = (unsigned short*)take((size_t)Btok * D * 2);
  unsigned short* h3_bf  = (unsigned short*)take((size_t)Btok * DFF * 2);

  // weight prep: fp32 [K][N] -> bf16 [N][K]
  dim3 tb(32, 8);
  transpose_f32_to_bf16<<<dim3(NQKV / 32, D / 32), tb, 0, stream>>>(
      qkv_w, qkvw_t, D, NQKV);
  transpose_f32_to_bf16<<<dim3(D / 32, D / 32), tb, 0, stream>>>(
      out_w, outw_t, D, D);
  transpose_f32_to_bf16<<<dim3(DFF / 32, D / 32), tb, 0, stream>>>(
      fc1_w, fc1w_t, D, DFF);
  transpose_f32_to_bf16<<<dim3(D / 32, DFF / 32), tb, 0, stream>>>(
      fc2_w, fc2w_t, DFF, D);

  // attention sublayer
  ln_bf16_kernel<<<Btok, 256, 0, stream>>>(x, ln1_g, ln1_b, h1_bf);
  gemm_bf16_kernel<<<dim3(NQKV / BN, Btok / BM), 256, 0, stream>>>(
      h1_bf, qkvw_t, qkv_b, nullptr, nullptr, qkv_bf, Btok, NQKV, D, 0);
  attn_kernel<<<2 * HEADS * (SEQ / 16), 32, 0, stream>>>(qkv_bf, mask, attn_bf);
  gemm_bf16_kernel<<<dim3(D / BN, Btok / BM), 256, 0, stream>>>(
      attn_bf, outw_t, out_b, x, x1, nullptr, Btok, D, D, 0);

  // MLP sublayer
  ln_bf16_kernel<<<Btok, 256, 0, stream>>>(x1, ln2_g, ln2_b, h2_bf);
  gemm_bf16_kernel<<<dim3(DFF / BN, Btok / BM), 256, 0, stream>>>(
      h2_bf, fc1w_t, fc1_b, nullptr, nullptr, h3_bf, Btok, DFF, D, 1);
  gemm_bf16_kernel<<<dim3(D / BN, Btok / BM), 256, 0, stream>>>(
      h3_bf, fc2w_t, fc2_b, x1, (float*)d_out, nullptr, Btok, D, DFF, 0);
}
